// RNN_3341484556758
// MI455X (gfx1250) — compile-verified, matrix operand order checked by reading, not run on hardware
//
#include <hip/hip_runtime.h>

// Problem constants (from reference): B=256, T=80, IN=39, H=256, L=5
#define B_   256
#define T_   80
#define IN_  39
#define INP_ 64          // IN padded to multiple of 32 for bf16 WMMA K
#define H_   256
#define G4_  1024        // 4*H gates
#define L_   5
#define M_   (B_ * T_)   // 20480 rows for the time-parallel GEMMs

typedef __attribute__((ext_vector_type(16))) __bf16 v16bf;
typedef __attribute__((ext_vector_type(8)))  __bf16 v8bf;
typedef __attribute__((ext_vector_type(8)))  float  v8f;
typedef int v4i __attribute__((vector_size(16)));   // matches builtin param type

#define AS1 __attribute__((address_space(1)))
#define AS3 __attribute__((address_space(3)))

// gfx1250 async global->LDS path (guarded so absence of the builtin cannot
// break the build; fallback = direct global loads + prefetch).
#if defined(__gfx1250__) && __has_builtin(__builtin_amdgcn_global_load_async_to_lds_b128)
#define HAVE_ASYNC 1
#else
#define HAVE_ASYNC 0
#endif

template <int N>
__device__ __forceinline__ void wait_asynccnt() {
#if __has_builtin(__builtin_amdgcn_s_wait_asynccnt)
    __builtin_amdgcn_s_wait_asynccnt(N);
#elif defined(__gfx1250__)
    asm volatile("s_wait_asynccnt %0" :: "i"(N) : "memory");
#endif
}

#if HAVE_ASYNC
// One 16-byte async DMA: global (AS1) -> LDS (AS3), ASYNCcnt-tracked.
__device__ __forceinline__ void async_copy_b128(const float* g, float* l) {
    __builtin_amdgcn_global_load_async_to_lds_b128(
        (AS1 v4i*)(AS1 void*)g,   // addrspacecast, then pointee reinterpret
        (AS3 v4i*)(AS3 void*)l,
        0, 0);                    // imm offset, imm cpol
}
#endif

// ---------------------------------------------------------------- helpers ---

__device__ __forceinline__ unsigned short f2bf(float f) {
    unsigned int u = __float_as_uint(f);
    u += 0x7fffu + ((u >> 16) & 1u);          // round-to-nearest-even
    return (unsigned short)(u >> 16);
}

__device__ __forceinline__ float bf2f(unsigned short b) {
    return __uint_as_float(((unsigned int)b) << 16);
}

__device__ __forceinline__ float sigm(float x) {
    return 1.0f / (1.0f + __expf(-x));        // v_exp_f32 (TRANS, co-executes with WMMA)
}

__device__ __forceinline__ float tanh_f(float x) {
    x = fminf(fmaxf(x, -15.0f), 15.0f);
    float e = __expf(2.0f * x);
    return (e - 1.0f) / (e + 1.0f);
}

// Load a 16x32 bf16 WMMA fragment from a row-major [rows, ld] bf16 matrix.
// ISA layout (CDNA5 ISA): lanes 0-15 hold row M=row0+lane, K = k0..k0+7 in
// VGPR0-3 and k0+16..k0+23 in VGPR4-7; lanes 16-31 hold the +8 K-halves.
// The SAME loader serves B-frags of (W^T) read from row-major W [N, K].
__device__ __forceinline__ v16bf load_frag_rm(const unsigned short* __restrict__ base,
                                              int ld, int row0, int k0,
                                              int lr, int lhi) {
    const unsigned short* p = base + (size_t)(row0 + lr) * ld + k0 + (lhi << 3);
    v8bf lo = *(const v8bf*)p;            // 16B -> global_load_b128
    v8bf hi = *(const v8bf*)(p + 16);     // 16B -> global_load_b128
    v16bf r;
#pragma unroll
    for (int i = 0; i < 8; ++i) { r[i] = lo[i]; r[i + 8] = hi[i]; }
    return r;
}

__device__ __forceinline__ v8f wmma_bf16(v16bf a, v16bf b, v8f c) {
    return __builtin_amdgcn_wmma_f32_16x16x32_bf16(false, a, false, b,
                                                   (short)0, c, false, false);
}

// ---------------------------------------------------------------- kernels ---

// fp32 [rows, cs] -> bf16 [rows, cd], zero-padded columns cs..cd-1.
__global__ void cvt_pad_kernel(const float* __restrict__ src,
                               unsigned short* __restrict__ dst,
                               int rows, int cs, int cd) {
    int idx = blockIdx.x * 256 + threadIdx.x;
    if (idx >= rows * cd) return;
    int r = idx / cd, c = idx - r * cd;
    dst[idx] = (c < cs) ? f2bf(src[(size_t)r * cs + c]) : (unsigned short)0;
}

// C[M_, 1024] = A[M_, K] @ W[1024, K]^T + (bih + bhh), fp32 accumulate.
// One wave per 16x16 C tile; 8 waves per block.
__global__ void __launch_bounds__(256, 2)
wmma_gemm_bias_kernel(const unsigned short* __restrict__ A,
                      const unsigned short* __restrict__ W,
                      const float* __restrict__ bih,
                      const float* __restrict__ bhh,
                      float* __restrict__ C, int K) {
    const int tile = blockIdx.x * 8 + (threadIdx.x >> 5);
    const int tm = tile >> 6;            // 1024/16 = 64 tiles along N
    const int tn = tile & 63;
    const int lane = threadIdx.x & 31, lr = lane & 15, lhi = lane >> 4;

    v8f acc = {};
#pragma unroll 8
    for (int k0 = 0; k0 < K; k0 += 32) {
        v16bf a = load_frag_rm(A, K, tm << 4, k0, lr, lhi);
        v16bf b = load_frag_rm(W, K, tn << 4, k0, lr, lhi);
        acc = wmma_bf16(a, b, acc);
    }
    // C/D layout: VGPR r holds (M = 16*tm + 8*lhi + r, N = 16*tn + lr)
    const int n = (tn << 4) + lr;
    const float bias = bih[n] + bhh[n];
    const int m0 = (tm << 4) + (lhi << 3);
#pragma unroll
    for (int r = 0; r < 8; ++r)
        C[(size_t)(m0 + r) * G4_ + n] = acc[r] + bias;
}

// Recurrent T-loop. Each block owns 16 batch rows (independent LSTM states),
// 8 waves x 32 H-columns each. h lives in LDS (bf16), c in VGPRs.
// Per step per wave: 2 x 8 x 4 = 64 v_wmma_f32_16x16x32_bf16.
// gfx1250: step t+1's 64KB gate-preactivation tile is double-buffered in LDS
// via GLOBAL_LOAD_ASYNC_TO_LDS_B128, overlapping the WMMA chain of step t.
__global__ void __launch_bounds__(256, 1)
lstm_rec_kernel(const float* __restrict__ xp,           // [B_, T_, 1024] gate preacts
                const unsigned short* __restrict__ whh, // [1024, 256] bf16 row-major
                unsigned short* __restrict__ hout) {    // [B_, T_, 256] bf16
    __shared__ unsigned short hsh[16][264];             // stride 264: conflict-free b128
#if HAVE_ASYNC
    __shared__ float xbuf[2][16 * 1024];                // 2 x 64KB (fits 320KB WGP LDS)
#endif
    const int tid  = threadIdx.x;
    const int wave = tid >> 5, lane = tid & 31;
    const int lr = lane & 15, lhi = lane >> 4;
    const int b0 = blockIdx.x << 4;                     // batch base
    const int jbase = wave << 5;                        // this wave's 32 H-columns

    for (int i = tid; i < 16 * 264; i += 256) (&hsh[0][0])[i] = 0;  // h0 = 0
    float cst[2][8];
#pragma unroll
    for (int jh = 0; jh < 2; ++jh)
#pragma unroll
        for (int r = 0; r < 8; ++r) cst[jh][r] = 0.0f;  // c0 = 0

#if HAVE_ASYNC
    // Issue the async copy of step-t's [16 x 1024] fp32 preactivation tile.
    // 4096 B128 chunks over 256 threads -> 16 chunks per thread, ASYNCcnt-tracked.
    auto issue_async = [&](int t, int tb) {
#pragma unroll
        for (int c = 0; c < 16; ++c) {
            const int chunk = c * 256 + tid;            // 0..4095
            const int m     = chunk >> 8;               // row 0..15
            const int col   = (chunk & 255) << 2;       // dword column, step 4
            async_copy_b128(xp + ((size_t)(b0 + m) * T_ + t) * G4_ + col,
                            &xbuf[tb][m * 1024 + col]);
        }
    };
    issue_async(0, 0);
#endif
    __syncthreads();

    for (int t = 0; t < T_; ++t) {
#if HAVE_ASYNC
        const int tb = t & 1;
        if (t + 1 < T_) { issue_async(t + 1, tb ^ 1); wait_asynccnt<16>(); }
        else            { wait_asynccnt<0>(); }
        __syncthreads();   // step-t tile (written by all waves' asyncs) visible
#endif
        unsigned short hreg[2][8];
#pragma unroll
        for (int jh = 0; jh < 2; ++jh) {
            const int j0 = jbase + (jh << 4);
            const int m0 = lhi << 3;

            // init accumulators with gate preactivations (C operand of WMMA)
            v8f acc[4];
#pragma unroll
            for (int g = 0; g < 4; ++g) {
#if HAVE_ASYNC
                const float* xr = &xbuf[tb][m0 * 1024 + (g << 8) + j0 + lr];
#pragma unroll
                for (int r = 0; r < 8; ++r)
                    acc[g][r] = xr[r * 1024];           // ds_load_b32
#else
                const float* xr = xp + ((size_t)(b0 + m0) * T_ + t) * G4_
                                     + (g << 8) + j0 + lr;
#pragma unroll
                for (int r = 0; r < 8; ++r)
                    acc[g][r] = xr[(size_t)r * T_ * G4_];
#endif
            }

            // gates += h_{t-1} @ W_hh^T  (K = 256)
#pragma unroll
            for (int k0 = 0; k0 < H_; k0 += 32) {
                const unsigned short* hp = &hsh[lr][k0 + (lhi << 3)];
                v8bf lo = *(const v8bf*)hp;            // ds_load_b128
                v8bf hi = *(const v8bf*)(hp + 16);     // ds_load_b128
                v16bf a;
#pragma unroll
                for (int i = 0; i < 8; ++i) { a[i] = lo[i]; a[i + 8] = hi[i]; }
#pragma unroll
                for (int g = 0; g < 4; ++g) {
                    v16bf bb = load_frag_rm(whh, H_, (g << 8) + j0, k0, lr, lhi);
                    acc[g] = wmma_bf16(a, bb, acc[g]);
                }
            }

            // activations: gate order (i, f, g, o)
#pragma unroll
            for (int r = 0; r < 8; ++r) {
                float ig = sigm(acc[0][r]);
                float fg = sigm(acc[1][r]);
                float gg = tanh_f(acc[2][r]);
                float og = sigm(acc[3][r]);
                float c  = fg * cst[jh][r] + ig * gg;
                cst[jh][r] = c;
                hreg[jh][r] = f2bf(og * tanh_f(c));
            }
        }

        __syncthreads();   // all waves finished reading h_{t-1} from LDS
#pragma unroll
        for (int jh = 0; jh < 2; ++jh) {
            const int n = jbase + (jh << 4) + lr;
#pragma unroll
            for (int r = 0; r < 8; ++r) {
                const int m = (lhi << 3) + r;
                hsh[m][n] = hreg[jh][r];                                 // ds_store_b16
                hout[((size_t)(b0 + m) * T_ + t) * H_ + n] = hreg[jh][r];
            }
        }
#if !HAVE_ASYNC
        if (t + 1 < T_) {  // warm L2/WGP$ for next step's preactivations
            const float* pf = xp + ((size_t)(b0 + lr) * T_ + (t + 1)) * G4_ + wave * 128;
            __builtin_prefetch(pf, 0, 1);              // global_prefetch_b8
        }
#endif
        __syncthreads();
    }
}

// out[b,t] = h_last[b,t,:] . fc_w + fc_b ; one wave per row, shuffle reduce.
__global__ void __launch_bounds__(256)
fc_kernel(const unsigned short* __restrict__ h, const float* __restrict__ fcw,
          const float* __restrict__ fcb, float* __restrict__ out) {
    const int row  = blockIdx.x * 8 + (threadIdx.x >> 5);
    const int lane = threadIdx.x & 31;
    float s = 0.0f;
#pragma unroll
    for (int k = lane; k < H_; k += 32)
        s += bf2f(h[(size_t)row * H_ + k]) * fcw[k];
#pragma unroll
    for (int off = 16; off > 0; off >>= 1)
        s += __shfl_xor(s, off, 32);
    if (lane == 0) out[row] = s + fcb[0];
}

// ------------------------------------------------------------------ host ---

extern "C" void kernel_launch(void* const* d_in, const int* in_sizes, int n_in,
                              void* d_out, int out_size, void* d_ws, size_t ws_size,
                              hipStream_t stream) {
    const float* x    = (const float*)d_in[0];
    const float* Wih0 = (const float*)d_in[1];  // [1024, 39]
    const float* Wihr = (const float*)d_in[2];  // [4, 1024, 256]
    const float* Whh  = (const float*)d_in[3];  // [5, 1024, 256]
    const float* bih  = (const float*)d_in[4];  // [5, 1024]
    const float* bhh  = (const float*)d_in[5];  // [5, 1024]
    const float* fcw  = (const float*)d_in[6];  // [256]
    const float* fcb  = (const float*)d_in[7];  // [1]
    float* out = (float*)d_out;

    char* ws = (char*)d_ws;
    size_t off = 0;
    auto take = [&](size_t bytes) -> void* {
        void* p = ws + off;
        off += (bytes + 255) & ~(size_t)255;
        return p;
    };
    unsigned short* xpad = (unsigned short*)take((size_t)M_ * INP_ * 2);
    unsigned short* wih0 = (unsigned short*)take((size_t)G4_ * INP_ * 2);
    unsigned short* wihr = (unsigned short*)take((size_t)4 * G4_ * H_ * 2);
    unsigned short* whh  = (unsigned short*)take((size_t)5 * G4_ * H_ * 2);
    float*          xprj = (float*)take((size_t)M_ * G4_ * 4);
    unsigned short* hA   = (unsigned short*)take((size_t)M_ * H_ * 2);
    unsigned short* hB   = (unsigned short*)take((size_t)M_ * H_ * 2);

    auto cdiv = [](long a, long b) { return (int)((a + b - 1) / b); };

    // bf16 conversion / padding (one-time per call, fully parallel)
    cvt_pad_kernel<<<cdiv((long)M_ * INP_, 256), 256, 0, stream>>>(x, xpad, M_, IN_, INP_);
    cvt_pad_kernel<<<cdiv((long)G4_ * INP_, 256), 256, 0, stream>>>(Wih0, wih0, G4_, IN_, INP_);
    cvt_pad_kernel<<<cdiv((long)4 * G4_ * H_, 256), 256, 0, stream>>>(Wihr, wihr, 4 * G4_, H_, H_);
    cvt_pad_kernel<<<cdiv((long)5 * G4_ * H_, 256), 256, 0, stream>>>(Whh, whh, 5 * G4_, H_, H_);

    const unsigned short* Acur = xpad;
    const unsigned short* Wcur = wih0;
    int K = INP_;
    unsigned short* hbuf[2] = {hA, hB};
    const int gemmGrid = (M_ / 16) * (G4_ / 16) / 8;  // 10240 blocks

    for (int l = 0; l < L_; ++l) {
        wmma_gemm_bias_kernel<<<gemmGrid, 256, 0, stream>>>(
            Acur, Wcur, bih + l * G4_, bhh + l * G4_, xprj, K);
        lstm_rec_kernel<<<B_ / 16, 256, 0, stream>>>(
            xprj, whh + (size_t)l * G4_ * H_, hbuf[l & 1]);
        Acur = hbuf[l & 1];
        Wcur = wihr + (size_t)l * G4_ * H_;  // next layer's input weights
        K = H_;
    }
    fc_kernel<<<M_ / 8, 256, 0, stream>>>(Acur, fcw, fcb, out);

    (void)in_sizes; (void)n_in; (void)out_size; (void)ws_size;
}
